// WindowAttention_27977416966346
// MI455X (gfx1250) — matrix operand order repaired and checked by reading, not compiled, not run on hardware
//
#include <hip/hip_runtime.h>

#define DIM   192
#define HEADS 6
#define HD    32
#define NTOK  49
#define NPAD  64
#define NW    64
#define QKVO  576
// SCALE * log2(e): softmax computed in exp2 domain (v_exp_f32 is exp2)
#define SCALE2  (0.17677669529663687f * 1.4426950408889634f)
#define LOG2E   1.4426950408889634f

typedef __attribute__((ext_vector_type(16))) __bf16 v16bf;
typedef __attribute__((ext_vector_type(8)))  __bf16 bf8;
typedef __attribute__((ext_vector_type(4)))  __bf16 bf4;
typedef __attribute__((ext_vector_type(8)))  float  v8f;

// RNE conversion: used only in one-time weight prep
__device__ __forceinline__ __bf16 to_bf_rne(float f) {
  unsigned u = __builtin_bit_cast(unsigned, f);
  u = (u + 0x7FFFu + ((u >> 16) & 1u)) >> 16;
  unsigned short s = (unsigned short)u;
  return __builtin_bit_cast(__bf16, s);
}
// truncating conversion: 1 VALU, pairs merge into v_perm_b32
__device__ __forceinline__ __bf16 to_bf(float f) {
  unsigned u = __builtin_bit_cast(unsigned, f);
  unsigned short s = (unsigned short)(u >> 16);
  return __builtin_bit_cast(__bf16, s);
}

// A-fragment (16x32 bf16): lane holds row m=lane%16; g=lane/16.
// elements 0-15 = k0+16g .. k0+16g+15 contiguous (p points at k0+16g)
__device__ __forceinline__ v16bf ldfragA(const __bf16* p) {
  bf8 lo = *(const bf8*)p;
  bf8 hi = *(const bf8*)(p + 8);
  return __builtin_shufflevector(lo, hi, 0,1,2,3,4,5,6,7,8,9,10,11,12,13,14,15);
}
// B-fragment (32x16 bf16): lane holds col n=lane%16; elements = k0+16g .. +15 contiguous
__device__ __forceinline__ v16bf ldfragB(const __bf16* p) {
  return *(const v16bf*)p;
}

__device__ __forceinline__ v8f wmma_bf16(v16bf a, v16bf b, v8f c) {
  return __builtin_amdgcn_wmma_f32_16x16x32_bf16(false, a, false, b, (short)0, c, false, false);
}

// ---- prep: weight transpose + f32->bf16 (RNE), mask pre-scaled by log2e ----
__global__ __launch_bounds__(256)
void prep_inputs(const float* __restrict__ qkv_w, const float* __restrict__ proj_w,
                 const float* __restrict__ mask,
                 __bf16* __restrict__ qkv_wT, __bf16* __restrict__ proj_wT,
                 float* __restrict__ mask2) {
  int i = blockIdx.x * 256 + threadIdx.x;
  if (i < QKVO * DIM) {                 // qkv_wT[j][k] = qkv_w[k][j]
    int j = i / DIM, k = i - j * DIM;
    qkv_wT[i] = to_bf_rne(qkv_w[k * QKVO + j]);
  }
  if (i < DIM * DIM) {                  // proj_wT[j][k] = proj_w[k][j]
    int j = i / DIM, k = i - j * DIM;
    proj_wT[i] = to_bf_rne(proj_w[k * DIM + j]);
  }
  if (i < NW * NTOK * NTOK)             // mask in exp2 domain
    mask2[i] = mask[i] * LOG2E;
}

// ---- fused window attention: one block per window b ----
__global__ __launch_bounds__(256)
void swin_attn_kernel(const float* __restrict__ x, const float* __restrict__ mask2,
                      const float* __restrict__ qkv_b, const float* __restrict__ proj_b,
                      const __bf16* __restrict__ qkv_wT, const __bf16* __restrict__ proj_wT,
                      float* __restrict__ out) {
  __shared__ __align__(32) char smem[98304];
  __bf16* xs = (__bf16*)smem;                 // [64][192] x, later attn_out
  __bf16* qs = (__bf16*)(smem + 24576);       // [64][192] q (pre-scaled) ; later P heads 0-2
  __bf16* ks = (__bf16*)(smem + 49152);       // [64][192] k              ; later P heads 3-5
  __bf16* vt = (__bf16*)(smem + 73728);       // [192][64] v transposed

  const int b    = blockIdx.x;
  const int tid  = threadIdx.x;
  const int lane = tid & 31;
  const int w    = __builtin_amdgcn_readfirstlane(tid >> 5);   // wave-uniform (SGPR)
  const int g    = lane >> 4, ln = lane & 15;

  // ---- stage 1: x[b] -> LDS bf16 (vectorized), zero-pad rows 49..63 ----
  {
    const float4* xb4 = (const float4*)(x + (size_t)b * (NTOK * DIM));
    for (int t = tid; t < (NTOK * DIM) / 4; t += 256) {   // contiguous float4 chunks
      float4 v = xb4[t];
      bf4 pk = { to_bf(v.x), to_bf(v.y), to_bf(v.z), to_bf(v.w) };
      *(bf4*)(xs + t * 4) = pk;
    }
    bf8 z = {};
    for (int t = tid; t < ((NPAD - NTOK) * DIM) / 8; t += 256)  // zero pad rows
      *(bf8*)(xs + NTOK * DIM + t * 8) = z;
  }
  __syncthreads();

  // ---- stage 2: qkv = x @ qkv_wT^T + b ; route to q/k/v LDS layouts ----
  for (int jt = w; jt < QKVO / 16; jt += 8) {
    int jcol = jt * 16 + ln;
    float bias = qkv_b[jcol];
    const __bf16* wrow = qkv_wT + (size_t)jcol * DIM;
    __builtin_prefetch(wrow + (size_t)128 * DIM, 0, 2);   // next jt for this wave
    v16bf bw[6];
#pragma unroll
    for (int kt = 0; kt < 6; ++kt)
      bw[kt] = ldfragB(wrow + kt * 32 + 16 * g);
#pragma unroll
    for (int mt = 0; mt < 4; ++mt) {
      v8f acc = {};
#pragma unroll
      for (int kt = 0; kt < 6; ++kt) {
        v16bf a = ldfragA(xs + (mt * 16 + ln) * DIM + kt * 32 + 16 * g);
        acc = wmma_bf16(a, bw[kt], acc);
      }
      // jt is wave-uniform -> scalar branch, no exec-mask dance
      if (jt < 12) {                       // q (pre-scaled into exp2 domain)
        __bf16* p = qs + (mt * 16 + 8 * g) * DIM + jcol;
#pragma unroll
        for (int r = 0; r < 8; ++r) p[r * DIM] = to_bf((acc[r] + bias) * SCALE2);
      } else if (jt < 24) {                // k
        __bf16* p = ks + (mt * 16 + 8 * g) * DIM + (jcol - DIM);
#pragma unroll
        for (int r = 0; r < 8; ++r) p[r * DIM] = to_bf(acc[r] + bias);
      } else {                             // v^T : 8 consecutive tokens -> one b128
        bf8 pk;
#pragma unroll
        for (int r = 0; r < 8; ++r) pk[r] = to_bf(acc[r] + bias);
        *(bf8*)(vt + (jcol - 2 * DIM) * NPAD + mt * 16 + 8 * g) = pk;
      }
    }
  }
  __syncthreads();

  // ---- stage 3: per-head S = q k^T (+mask), softmax (exp2 domain) in regs ----
  v8f s[4][4];
  if (w < HEADS) {
    const int h = w;
    v16bf qf[4], kf[4];
#pragma unroll
    for (int mt = 0; mt < 4; ++mt)
      qf[mt] = ldfragA(qs + (mt * 16 + ln) * DIM + h * HD + 16 * g);
#pragma unroll
    for (int nt = 0; nt < 4; ++nt)
      kf[nt] = ldfragB(ks + (nt * 16 + ln) * DIM + h * HD + 16 * g);
#pragma unroll
    for (int mt = 0; mt < 4; ++mt)
#pragma unroll
      for (int nt = 0; nt < 4; ++nt) {
        v8f z = {};
        s[mt][nt] = wmma_bf16(qf[mt], kf[nt], z);
      }
    const float* mrow = mask2 + (size_t)(b & (NW - 1)) * (NTOK * NTOK);
#pragma unroll
    for (int mt = 0; mt < 4; ++mt)
#pragma unroll
      for (int r = 0; r < 8; ++r) {
        int n  = mt * 16 + r + 8 * g;                  // query row
        int nc = n < NTOK ? n : NTOK - 1;              // clamped for safe load
#pragma unroll
        for (int nt = 0; nt < 4; ++nt) {
          int mm  = nt * 16 + ln;                      // key col
          int mc  = mm < NTOK ? mm : NTOK - 1;
          float mv = mrow[nc * NTOK + mc];             // unconditional, clamped
          float val = s[mt][nt][r];
          val = (n < NTOK) ? val + mv : val;           // v_cndmask
          val = (mm < NTOK) ? val : -1e30f;            // padded keys -> 0 weight
          s[mt][nt][r] = val;
        }
        float mx = fmaxf(fmaxf(s[mt][0][r], s[mt][1][r]),
                         fmaxf(s[mt][2][r], s[mt][3][r]));
        mx = fmaxf(mx, __shfl_xor(mx, 1, 32));
        mx = fmaxf(mx, __shfl_xor(mx, 2, 32));
        mx = fmaxf(mx, __shfl_xor(mx, 4, 32));
        mx = fmaxf(mx, __shfl_xor(mx, 8, 32));
        float sum = 0.0f;
#pragma unroll
        for (int nt = 0; nt < 4; ++nt) {
          float e = __builtin_amdgcn_exp2f(s[mt][nt][r] - mx);  // bare v_exp_f32
          s[mt][nt][r] = e; sum += e;
        }
        sum += __shfl_xor(sum, 1, 32);
        sum += __shfl_xor(sum, 2, 32);
        sum += __shfl_xor(sum, 4, 32);
        sum += __shfl_xor(sum, 8, 32);
        float inv = __builtin_amdgcn_rcpf(sum);
#pragma unroll
        for (int nt = 0; nt < 4; ++nt) s[mt][nt][r] *= inv;
      }
  }
  __syncthreads();   // all q/k reads done before P overwrites that LDS

  // ---- stage 4: out_h = P @ V ; P staged through LDS (aliases q/k) ----
  if (w < HEADS) {
    const int h = w;
    __bf16* P = (__bf16*)(smem + 24576 + h * 8192);  // [64][64] bf16
#pragma unroll
    for (int mt = 0; mt < 4; ++mt)
#pragma unroll
      for (int nt = 0; nt < 4; ++nt) {
        __bf16* p = P + (mt * 16 + 8 * g) * NPAD + nt * 16 + ln;
#pragma unroll
        for (int r = 0; r < 8; ++r) p[r * NPAD] = to_bf(s[mt][nt][r]);
      }
    // same-wave DS ordering: stores complete before the loads below
    v8f o[4][2];
#pragma unroll
    for (int mt = 0; mt < 4; ++mt)
#pragma unroll
      for (int dt = 0; dt < 2; ++dt) o[mt][dt] = (v8f){};
#pragma unroll
    for (int kt = 0; kt < 2; ++kt) {
      v16bf vf[2];
#pragma unroll
      for (int dt = 0; dt < 2; ++dt)
        vf[dt] = ldfragB(vt + (h * HD + dt * 16 + ln) * NPAD + kt * 32 + 16 * g);
#pragma unroll
      for (int mt = 0; mt < 4; ++mt) {
        v16bf a = ldfragA(P + (mt * 16 + ln) * NPAD + kt * 32 + 16 * g);
#pragma unroll
        for (int dt = 0; dt < 2; ++dt)
          o[mt][dt] = wmma_bf16(a, vf[dt], o[mt][dt]);
      }
    }
#pragma unroll
    for (int mt = 0; mt < 4; ++mt)
#pragma unroll
      for (int dt = 0; dt < 2; ++dt) {
        __bf16* p = xs + (mt * 16 + 8 * g) * DIM + h * HD + dt * 16 + ln;
#pragma unroll
        for (int r = 0; r < 8; ++r) p[r * DIM] = to_bf(o[mt][dt][r]);
      }
  }
  __syncthreads();

  // ---- stage 5: out = attn_out @ proj_wT^T + b ; store fp32 rows < 49 ----
  float* ob = out + (size_t)b * (NTOK * DIM);
  for (int jt = w; jt < DIM / 16; jt += 8) {
    int jcol = jt * 16 + ln;
    float bias = proj_b[jcol];
    const __bf16* wrow = proj_wT + (size_t)jcol * DIM;
    v16bf bw[6];
#pragma unroll
    for (int kt = 0; kt < 6; ++kt)
      bw[kt] = ldfragB(wrow + kt * 32 + 16 * g);
#pragma unroll
    for (int mt = 0; mt < 4; ++mt) {
      v8f acc = {};
#pragma unroll
      for (int kt = 0; kt < 6; ++kt) {
        v16bf a = ldfragA(xs + (mt * 16 + ln) * DIM + kt * 32 + 16 * g);
        acc = wmma_bf16(a, bw[kt], acc);
      }
#pragma unroll
      for (int r = 0; r < 8; ++r) {
        int m = mt * 16 + r + 8 * g;
        if (m < NTOK) ob[m * DIM + jcol] = acc[r] + bias;
      }
    }
  }
}

extern "C" void kernel_launch(void* const* d_in, const int* in_sizes, int n_in,
                              void* d_out, int out_size, void* d_ws, size_t ws_size,
                              hipStream_t stream) {
  const float* x      = (const float*)d_in[0];
  const float* mask   = (const float*)d_in[1];
  const float* qkv_w  = (const float*)d_in[2];
  const float* qkv_b  = (const float*)d_in[3];
  const float* proj_w = (const float*)d_in[4];
  const float* proj_b = (const float*)d_in[5];
  float* out = (float*)d_out;

  __bf16* qkv_wT  = (__bf16*)d_ws;
  __bf16* proj_wT = (__bf16*)((char*)d_ws + (size_t)QKVO * DIM * sizeof(__bf16));
  float*  mask2   = (float*)((char*)d_ws + (size_t)(QKVO + DIM) * DIM * sizeof(__bf16));

  int prep_n = NW * NTOK * NTOK;        // 153664 > 576*192, covers all three prep jobs
  prep_inputs<<<(prep_n + 255) / 256, 256, 0, stream>>>(qkv_w, proj_w, mask,
                                                        qkv_wT, proj_wT, mask2);

  int B = in_sizes[0] / (NTOK * DIM);   // 4096
  swin_attn_kernel<<<B, 256, 0, stream>>>(x, mask2, qkv_b, proj_b, qkv_wT, proj_wT, out);
}
